// GraphFormerEncoderLayer_18691697672318
// MI455X (gfx1250) — compile-verified, matrix operand order checked by reading, not run on hardware
//
#include <hip/hip_runtime.h>

// ---------------------------------------------------------------------------
// GraphFormer encoder layer for MI455X (gfx1250, wave32, WMMA).
// All GEMMs use v_wmma_f32_16x16x32_bf16 (bf16 inputs, f32 accumulate).
// Roofline: ~120 GFLOP of GEMM vs ~400MB of traffic (256MB of it the
// mandatory attn output) -> must be on the matrix pipes; bf16 weights
// (24MB) live entirely in the 192MB L2.
// Register blocking: each wave owns a 32x64 C-tile (2x4 f32 accumulators),
// so one k-step costs 12 b128 loads for 8 WMMAs (1.5 loads/WMMA).
// ---------------------------------------------------------------------------

typedef __bf16 bf16_t;
typedef __attribute__((ext_vector_type(16))) __bf16 v16bf;
typedef __attribute__((ext_vector_type(8)))  __bf16 v8bf;
typedef __attribute__((ext_vector_type(8)))  float  v8f;
typedef __attribute__((ext_vector_type(4)))  float  v4f;

static constexpr int Bn  = 4;
static constexpr int Tn  = 1024;
static constexpr int Dn  = 1024;
static constexpr int Hn  = 16;
static constexpr int Fn  = 4096;
static constexpr int DHn = 64;      // Dn / Hn
static constexpr int NTn = Bn * Tn; // 4096 token rows
static constexpr float EPSn = 1e-5f;

// ---------------------------------------------------------------------------
// WMMA helpers.
// 16-bit A/B operand layout (ISA 7.12.2): lane l holds matrix row (l & 15);
// v16bf elements 0..7  = K  kb..kb+7,
//                8..15 = K  kb+16..kb+23,  where kb = k0 + 8*(l >= 16).
// So each lane loads two contiguous 16-byte chunks -> global_load_b128.
// ---------------------------------------------------------------------------
__device__ __forceinline__ v16bf load_op_bf16(const bf16_t* __restrict__ p) {
  v8bf lo = *reinterpret_cast<const v8bf*>(p);
  v8bf hi = *reinterpret_cast<const v8bf*>(p + 16);
  v16bf r;
#pragma unroll
  for (int e = 0; e < 8; ++e) { r[e] = lo[e]; r[e + 8] = hi[e]; }
  return r;
}

// Same operand but source data is f32 (attention probabilities): convert in-reg.
__device__ __forceinline__ v16bf load_op_f32(const float* __restrict__ p) {
  v4f l0 = *reinterpret_cast<const v4f*>(p);
  v4f l1 = *reinterpret_cast<const v4f*>(p + 4);
  v4f h0 = *reinterpret_cast<const v4f*>(p + 16);
  v4f h1 = *reinterpret_cast<const v4f*>(p + 20);
  v16bf r;
#pragma unroll
  for (int e = 0; e < 4; ++e) {
    r[e]      = (bf16_t)l0[e];
    r[4 + e]  = (bf16_t)l1[e];
    r[8 + e]  = (bf16_t)h0[e];
    r[12 + e] = (bf16_t)h1[e];
  }
  return r;
}

__device__ __forceinline__ v8f wmma_bf16(v16bf a, v16bf b, v8f c) {
  // (neg_a, A, neg_b, B, c_mod, C, reuse_a, reuse_b)
  return __builtin_amdgcn_wmma_f32_16x16x32_bf16(false, a, false, b,
                                                 (short)0, c, false, false);
}

// C/D 16x16 f32 layout: lane n = (l & 15); VGPR g -> row (l>=16 ? 8 : 0) + g.
__device__ __forceinline__ void gemm_store_tile(
    v8f acc, int mtile, int ntile, int mb, int cn,
    const float* __restrict__ bias, const float* __restrict__ Res, int ldr,
    float* __restrict__ Cf, bf16_t* __restrict__ Cbf, int ldc,
    float scale, int gelu_flag)
{
  int n = ntile + cn;
  float bv = bias ? bias[n] : 0.0f;
#pragma unroll
  for (int g = 0; g < 8; ++g) {
    int m = mtile + mb + g;
    float v = (acc[g] + bv) * scale;
    if (gelu_flag) {  // tanh-approx GELU (jax.nn.gelu default)
      float t = 0.7978845608028654f * (v + 0.044715f * v * v * v);
      v = 0.5f * v * (1.0f + tanhf(t));
    }
    if (Res) v += Res[(long long)m * ldr + n];
    long long o = (long long)m * ldc + n;
    if (Cf)  Cf[o]  = v;
    if (Cbf) Cbf[o] = (bf16_t)v;
  }
}

// ---------------------------------------------------------------------------
// Generic WMMA GEMM: C[M,N] = epilogue(A[M,K] * BT[N,K]^T).
// Block = 128 threads = 4 waves stacked along M; each wave owns a 32x64 tile
// (2x4 WMMA accumulators), so the block covers 128x64.
// Requires M % 128 == 0, N % 64 == 0, K % 32 == 0 (true for every call here).
// ---------------------------------------------------------------------------
__global__ void __launch_bounds__(128) gf_gemm_bf16_wmma(
    const bf16_t* __restrict__ A, int lda,
    const bf16_t* __restrict__ BT, int ldb,
    const float* __restrict__ bias,
    const float* __restrict__ Res, int ldr,
    float* __restrict__ Cf, bf16_t* __restrict__ Cbf, int ldc,
    int M, int N, int K, float scale, int gelu_flag)
{
  int lane = threadIdx.x & 31;
  int wid  = threadIdx.x >> 5;
  int m0 = blockIdx.y * 128 + wid * 32;
  int n0 = blockIdx.x * 64;
  if (m0 >= M || n0 >= N) return;

  int arow = lane & 15;
  int koff = (lane >> 4) << 3;

  const bf16_t* pa0 = A + (long long)(m0 + arow) * lda + koff;
  const bf16_t* pa1 = pa0 + 16LL * lda;
  const bf16_t* pb[4];
#pragma unroll
  for (int j = 0; j < 4; ++j)
    pb[j] = BT + (long long)(n0 + 16 * j + arow) * ldb + koff;

  v8f acc0[4] = {{}, {}, {}, {}};
  v8f acc1[4] = {{}, {}, {}, {}};

  for (int k = 0; k < K; k += 32) {
    if (k + 128 < K) {               // global_prefetch_b8 four k-steps ahead
      __builtin_prefetch(pa0 + k + 128, 0, 1);
      __builtin_prefetch(pa1 + k + 128, 0, 1);
      __builtin_prefetch(pb[0] + k + 128, 0, 1);
      __builtin_prefetch(pb[2] + k + 128, 0, 1);
    }
    v16bf a0 = load_op_bf16(pa0 + k);
    v16bf a1 = load_op_bf16(pa1 + k);
#pragma unroll
    for (int j = 0; j < 4; ++j) {
      v16bf b = load_op_bf16(pb[j] + k);
      acc0[j] = wmma_bf16(a0, b, acc0[j]);
      acc1[j] = wmma_bf16(a1, b, acc1[j]);
    }
  }

  int cn = lane & 15;
  int mb = (lane >> 4) * 8;
#pragma unroll
  for (int j = 0; j < 4; ++j) {
    gemm_store_tile(acc0[j], m0,      n0 + 16 * j, mb, cn, bias, Res, ldr, Cf, Cbf, ldc, scale, gelu_flag);
    gemm_store_tile(acc1[j], m0 + 16, n0 + 16 * j, mb, cn, bias, Res, ldr, Cf, Cbf, ldc, scale, gelu_flag);
  }
}

// ---------------------------------------------------------------------------
// Attention scores: S[z, t, s] = sum_dh q[b,t,h,dh] * kb[b,s,h,dh]
// (1/sqrt(DH) is pre-folded into q).  z = b*H + h, K = DH = 64 (2 k-steps),
// so this kernel is store-bound; keep the lighter 2x2 wave tile.
// ---------------------------------------------------------------------------
__global__ void __launch_bounds__(128) gf_scores_wmma(
    const bf16_t* __restrict__ Q,   // (B*T, D)
    const bf16_t* __restrict__ KB,  // (B*T, D)
    float* __restrict__ S)          // (B*H, T, T)
{
  int z = blockIdx.z;
  int b = z / Hn, h = z % Hn;
  const bf16_t* q  = Q  + (long long)b * Tn * Dn + h * DHn;
  const bf16_t* kb = KB + (long long)b * Tn * Dn + h * DHn;
  float* s = S + (long long)z * Tn * Tn;

  int lane = threadIdx.x & 31;
  int wid  = threadIdx.x >> 5;
  int m0 = blockIdx.y * 64 + (wid >> 1) * 32;  // t
  int n0 = blockIdx.x * 64 + (wid & 1) * 32;   // s

  int arow = lane & 15;
  int koff = (lane >> 4) << 3;
  const bf16_t* pa0 = q  + (long long)(m0 + arow) * Dn + koff;
  const bf16_t* pa1 = pa0 + 16LL * Dn;
  const bf16_t* pb0 = kb + (long long)(n0 + arow) * Dn + koff;
  const bf16_t* pb1 = pb0 + 16LL * Dn;

  v8f acc00 = {}, acc01 = {}, acc10 = {}, acc11 = {};
#pragma unroll
  for (int k = 0; k < DHn; k += 32) {
    v16bf a0 = load_op_bf16(pa0 + k);
    v16bf a1 = load_op_bf16(pa1 + k);
    v16bf b0 = load_op_bf16(pb0 + k);
    v16bf b1 = load_op_bf16(pb1 + k);
    acc00 = wmma_bf16(a0, b0, acc00);
    acc01 = wmma_bf16(a0, b1, acc01);
    acc10 = wmma_bf16(a1, b0, acc10);
    acc11 = wmma_bf16(a1, b1, acc11);
  }

  int cn = lane & 15;
  int mb = (lane >> 4) * 8;
#pragma unroll
  for (int g = 0; g < 8; ++g) {
    s[(long long)(m0 + mb + g)      * Tn + n0 + cn]      = acc00[g];
    s[(long long)(m0 + mb + g)      * Tn + n0 + 16 + cn] = acc01[g];
    s[(long long)(m0 + 16 + mb + g) * Tn + n0 + cn]      = acc10[g];
    s[(long long)(m0 + 16 + mb + g) * Tn + n0 + 16 + cn] = acc11[g];
  }
}

// ---------------------------------------------------------------------------
// ctx[b,t,h,:] = sum_s attn[b,h,t,s] * v[b,s,h,:]
// A = attn (f32 -> bf16 in-reg), B^T = V^T[z] (DH x T, contiguous over s).
// 2x4 wave tile: the 4 B-tiles cover DH=64 exactly.
// ---------------------------------------------------------------------------
__global__ void __launch_bounds__(128) gf_av_wmma(
    const float* __restrict__ Attn,  // (B*H, T, T)
    const bf16_t* __restrict__ VT,   // (B*H, DH, T)
    bf16_t* __restrict__ Ctx)        // (B*T, D)
{
  int z = blockIdx.z;
  int b = z / Hn, h = z % Hn;
  const float*  at = Attn + (long long)z * Tn * Tn;
  const bf16_t* vt = VT   + (long long)z * DHn * Tn;
  bf16_t* out = Ctx + (long long)b * Tn * Dn + h * DHn;

  int lane = threadIdx.x & 31;
  int wid  = threadIdx.x >> 5;
  int m0 = blockIdx.y * 128 + wid * 32;  // t

  int arow = lane & 15;
  int koff = (lane >> 4) << 3;
  const float* pa0 = at + (long long)(m0 + arow) * Tn + koff;
  const float* pa1 = pa0 + 16LL * Tn;
  const bf16_t* pb[4];
#pragma unroll
  for (int j = 0; j < 4; ++j)
    pb[j] = vt + (long long)(16 * j + arow) * Tn + koff;

  v8f acc0[4] = {{}, {}, {}, {}};
  v8f acc1[4] = {{}, {}, {}, {}};

  for (int k = 0; k < Tn; k += 32) {
    v16bf a0 = load_op_f32(pa0 + k);
    v16bf a1 = load_op_f32(pa1 + k);
#pragma unroll
    for (int j = 0; j < 4; ++j) {
      v16bf bj = load_op_bf16(pb[j] + k);
      acc0[j] = wmma_bf16(a0, bj, acc0[j]);
      acc1[j] = wmma_bf16(a1, bj, acc1[j]);
    }
  }

  int cn = lane & 15;
  int mb = (lane >> 4) * 8;
#pragma unroll
  for (int j = 0; j < 4; ++j) {
#pragma unroll
    for (int g = 0; g < 8; ++g) {
      out[(long long)(m0 + mb + g)      * Dn + 16 * j + cn] = (bf16_t)acc0[j][g];
      out[(long long)(m0 + 16 + mb + g) * Dn + 16 * j + cn] = (bf16_t)acc1[j][g];
    }
  }
}

// ---------------------------------------------------------------------------
// LayerNorm: one 256-thread block per row of ncols (=1024) f32 values.
// Emits f32 (residual path) and optionally bf16 (GEMM input path).
// ---------------------------------------------------------------------------
__global__ void __launch_bounds__(256) gf_layernorm(
    const float* __restrict__ X, const float* __restrict__ gamma,
    const float* __restrict__ beta, float* __restrict__ Yf,
    bf16_t* __restrict__ Ybf, int ncols)
{
  __shared__ float sh_s[256], sh_q[256];
  long long row = blockIdx.x;
  const float* x = X + row * (long long)ncols;
  int tid = threadIdx.x;

  float s = 0.f, s2 = 0.f;
  for (int c = tid; c < ncols; c += 256) { float v = x[c]; s += v; s2 += v * v; }
  sh_s[tid] = s; sh_q[tid] = s2; __syncthreads();
  for (int o = 128; o > 0; o >>= 1) {
    if (tid < o) { sh_s[tid] += sh_s[tid + o]; sh_q[tid] += sh_q[tid + o]; }
    __syncthreads();
  }
  float mean = sh_s[0] / (float)ncols;
  float var  = sh_q[0] / (float)ncols - mean * mean;
  float rstd = rsqrtf(var + EPSn);

  for (int c = tid; c < ncols; c += 256) {
    float v = (x[c] - mean) * rstd * gamma[c] + beta[c];
    if (Yf)  Yf[row * (long long)ncols + c]  = v;
    if (Ybf) Ybf[row * (long long)ncols + c] = (bf16_t)v;
  }
}

// ---------------------------------------------------------------------------
// Row softmax over 1024 cols, in place, LDS-staged (one block per row).
// ---------------------------------------------------------------------------
__global__ void __launch_bounds__(256) gf_softmax_rows(float* __restrict__ A)
{
  __shared__ float buf[Tn];
  __shared__ float red[256];
  long long row = blockIdx.x;
  float* a = A + row * (long long)Tn;
  int tid = threadIdx.x;

  float mx = -3.4e38f;
  for (int c = tid; c < Tn; c += 256) { float v = a[c]; buf[c] = v; mx = fmaxf(mx, v); }
  red[tid] = mx; __syncthreads();
  for (int o = 128; o > 0; o >>= 1) {
    if (tid < o) red[tid] = fmaxf(red[tid], red[tid + o]);
    __syncthreads();
  }
  mx = red[0]; __syncthreads();

  float s = 0.f;
  for (int c = tid; c < Tn; c += 256) { float e = __expf(buf[c] - mx); buf[c] = e; s += e; }
  red[tid] = s; __syncthreads();
  for (int o = 128; o > 0; o >>= 1) {
    if (tid < o) red[tid] += red[tid + o];
    __syncthreads();
  }
  float inv = 1.0f / red[0];
  for (int c = tid; c < Tn; c += 256) a[c] = buf[c] * inv;
}

// ---------------------------------------------------------------------------
// Small elementwise kernels.
// ---------------------------------------------------------------------------
// fp32 W[K][N] -> bf16 WT[N][K]  (coalesced reads)
__global__ void __launch_bounds__(256) gf_transpose_cvt(
    const float* __restrict__ W, bf16_t* __restrict__ WT, int Kd, int Nd)
{
  long long i = (long long)blockIdx.x * 256 + threadIdx.x;
  if (i >= (long long)Kd * Nd) return;
  int n = (int)(i % Nd);
  int k = (int)(i / Nd);
  WT[(long long)n * Kd + k] = (bf16_t)W[i];
}

// kb += spatial + edge (bf16 in place; encodings reshape (B,T,D)->(B,T,H,DH) is flat)
__global__ void __launch_bounds__(256) gf_add_enc(
    bf16_t* __restrict__ kb, const float* __restrict__ sp,
    const float* __restrict__ ed, long long n)
{
  long long i = (long long)blockIdx.x * 256 + threadIdx.x;
  if (i < n) kb[i] = (bf16_t)((float)kb[i] + sp[i] + ed[i]);
}

// v (B*T, D) -> vT (B*H, DH, T)
__global__ void __launch_bounds__(256) gf_transpose_v(
    const bf16_t* __restrict__ Vn, bf16_t* __restrict__ VT)
{
  long long i = (long long)blockIdx.x * 256 + threadIdx.x;
  long long total = (long long)Bn * Hn * DHn * Tn;
  if (i >= total) return;
  int s = (int)(i % Tn);
  long long r = i / Tn;
  int dh = (int)(r % DHn);
  int bh = (int)(r / DHn);
  int b = bh / Hn, h = bh % Hn;
  VT[i] = Vn[((long long)(b * Tn + s)) * Dn + h * DHn + dh];
}

// ---------------------------------------------------------------------------
// Orchestration
// ---------------------------------------------------------------------------
extern "C" void kernel_launch(void* const* d_in, const int* in_sizes, int n_in,
                              void* d_out, int out_size, void* d_ws, size_t ws_size,
                              hipStream_t stream) {
  (void)in_sizes; (void)n_in; (void)out_size; (void)ws_size;

  const float* x    = (const float*)d_in[0];
  const float* sp   = (const float*)d_in[1];
  const float* ed   = (const float*)d_in[2];
  const float* ln1g = (const float*)d_in[3];
  const float* ln1b = (const float*)d_in[4];
  const float* Wq   = (const float*)d_in[5];
  const float* bq   = (const float*)d_in[6];
  const float* Wk   = (const float*)d_in[7];
  const float* bk   = (const float*)d_in[8];
  const float* Wv   = (const float*)d_in[9];
  const float* bv   = (const float*)d_in[10];
  const float* Wo   = (const float*)d_in[11];
  const float* bo   = (const float*)d_in[12];
  const float* ln2g = (const float*)d_in[13];
  const float* ln2b = (const float*)d_in[14];
  const float* W1   = (const float*)d_in[15];
  const float* b1   = (const float*)d_in[16];
  const float* W2   = (const float*)d_in[17];
  const float* b2   = (const float*)d_in[18];
  const float* lnbg = (const float*)d_in[19];
  const float* lnbb = (const float*)d_in[20];

  float* out_x = (float*)d_out;                            // (B,T,D)
  float* attn  = (float*)d_out + (long long)NTn * Dn;      // (B,H,T,T)

  // Workspace carve-up (~160 MB).
  char* w = (char*)d_ws;
  size_t off = 0;
  auto alloc = [&](size_t bytes) {
    size_t o = (off + 255) & ~(size_t)255;
    off = o + bytes;
    return o;
  };
  const size_t ND  = (size_t)NTn * Dn;
  const size_t NF  = (size_t)NTn * Fn;
  float*  xln_f  = (float*)(w + alloc(ND * 4));  // post-LN1 x (residual); reused as FFN2 out
  float*  ctxp_f = (float*)(w + alloc(ND * 4));  // out-proj + residual (pre-LN2)
  float*  x2_f   = (float*)(w + alloc(ND * 4));  // post-LN2 (FFN residual)
  bf16_t* xln_bf = (bf16_t*)(w + alloc(ND * 2));
  bf16_t* x2_bf  = (bf16_t*)(w + alloc(ND * 2));
  bf16_t* q_bf   = (bf16_t*)(w + alloc(ND * 2));
  bf16_t* kb_bf  = (bf16_t*)(w + alloc(ND * 2));
  bf16_t* v_bf   = (bf16_t*)(w + alloc(ND * 2));
  bf16_t* vT_bf  = (bf16_t*)(w + alloc(ND * 2));
  bf16_t* ctx_bf = (bf16_t*)(w + alloc(ND * 2));
  bf16_t* h_bf   = (bf16_t*)(w + alloc(NF * 2));
  bf16_t* WqT = (bf16_t*)(w + alloc((size_t)Dn * Dn * 2));
  bf16_t* WkT = (bf16_t*)(w + alloc((size_t)Dn * Dn * 2));
  bf16_t* WvT = (bf16_t*)(w + alloc((size_t)Dn * Dn * 2));
  bf16_t* WoT = (bf16_t*)(w + alloc((size_t)Dn * Dn * 2));
  bf16_t* W1T = (bf16_t*)(w + alloc((size_t)Dn * Fn * 2));
  bf16_t* W2T = (bf16_t*)(w + alloc((size_t)Fn * Dn * 2));

  // 1) Weights -> bf16, transposed so both WMMA operands load contiguously.
  int gDD = (Dn * Dn + 255) / 256, gDF = (Dn * Fn + 255) / 256;
  gf_transpose_cvt<<<gDD, 256, 0, stream>>>(Wq, WqT, Dn, Dn);
  gf_transpose_cvt<<<gDD, 256, 0, stream>>>(Wk, WkT, Dn, Dn);
  gf_transpose_cvt<<<gDD, 256, 0, stream>>>(Wv, WvT, Dn, Dn);
  gf_transpose_cvt<<<gDD, 256, 0, stream>>>(Wo, WoT, Dn, Dn);
  gf_transpose_cvt<<<gDF, 256, 0, stream>>>(W1, W1T, Dn, Fn);
  gf_transpose_cvt<<<gDF, 256, 0, stream>>>(W2, W2T, Fn, Dn);

  // 2) LN1 -> xln (f32 residual + bf16 GEMM input)
  gf_layernorm<<<NTn, 256, 0, stream>>>(x, ln1g, ln1b, xln_f, xln_bf, Dn);

  // 3) Q,K,V projections (1/sqrt(DH)=0.125 folded into Q)
  dim3 blk(128);
  dim3 gQKV(Dn / 64, NTn / 128);   // 64-wide, 128-tall C blocks
  gf_gemm_bf16_wmma<<<gQKV, blk, 0, stream>>>(xln_bf, Dn, WqT, Dn, bq, nullptr, 0,
                                              nullptr, q_bf, Dn, NTn, Dn, Dn, 0.125f, 0);
  gf_gemm_bf16_wmma<<<gQKV, blk, 0, stream>>>(xln_bf, Dn, WkT, Dn, bk, nullptr, 0,
                                              nullptr, kb_bf, Dn, NTn, Dn, Dn, 1.0f, 0);
  gf_gemm_bf16_wmma<<<gQKV, blk, 0, stream>>>(xln_bf, Dn, WvT, Dn, bv, nullptr, 0,
                                              nullptr, v_bf, Dn, NTn, Dn, Dn, 1.0f, 0);

  // 4) K-bias: kb = k + spatial + edge ; V transposed for the AV GEMM.
  long long nel = (long long)NTn * Dn;
  gf_add_enc<<<(int)((nel + 255) / 256), 256, 0, stream>>>(kb_bf, sp, ed, nel);
  gf_transpose_v<<<(int)((nel + 255) / 256), 256, 0, stream>>>(v_bf, vT_bf);

  // 5) scores -> softmax (f32, written straight into the attn output region).
  gf_scores_wmma<<<dim3(Tn / 64, Tn / 64, Bn * Hn), blk, 0, stream>>>(q_bf, kb_bf, attn);
  gf_softmax_rows<<<Bn * Hn * Tn, 256, 0, stream>>>(attn);

  // 6) ctx = attn @ V
  gf_av_wmma<<<dim3(1, Tn / 128, Bn * Hn), blk, 0, stream>>>(attn, vT_bf, ctx_bf);

  // 7) out-proj + residual -> LN2
  gf_gemm_bf16_wmma<<<gQKV, blk, 0, stream>>>(ctx_bf, Dn, WoT, Dn, bo, xln_f, Dn,
                                              ctxp_f, nullptr, Dn, NTn, Dn, Dn, 1.0f, 0);
  gf_layernorm<<<NTn, 256, 0, stream>>>(ctxp_f, ln2g, ln2b, x2_f, x2_bf, Dn);

  // 8) FFN: gelu(x2 @ W1 + b1) @ W2 + b2 + x2 -> LNb -> d_out.x
  gf_gemm_bf16_wmma<<<dim3(Fn / 64, NTn / 128), blk, 0, stream>>>(
      x2_bf, Dn, W1T, Dn, b1, nullptr, 0, nullptr, h_bf, Fn, NTn, Fn, Dn, 1.0f, 1);
  gf_gemm_bf16_wmma<<<gQKV, blk, 0, stream>>>(h_bf, Fn, W2T, Fn, b2, x2_f, Dn,
                                              xln_f /*reuse*/, nullptr, Dn, NTn, Dn, Fn, 1.0f, 0);
  gf_layernorm<<<NTn, 256, 0, stream>>>(xln_f, lnbg, lnbb, out_x, nullptr, Dn);
}
